// CutsSelector_44470091383035
// MI455X (gfx1250) — matrix operand order repaired and checked by reading, not compiled, not run on hardware
//
#include <hip/hip_runtime.h>
#include <hip/hip_bf16.h>

typedef __attribute__((ext_vector_type(16))) __bf16 v16bf;
typedef __attribute__((ext_vector_type(8)))  float  v8f;
typedef __attribute__((ext_vector_type(8)))  unsigned short ushort8;

union BF16x16 {
    v16bf   v;
    ushort8 h[2];
};

__device__ __forceinline__ unsigned short f32_to_bf16(float f) {
    unsigned int u = __float_as_uint(f);
    unsigned int r = 0x7FFFu + ((u >> 16) & 1u);
    return (unsigned short)((u + r) >> 16);
}

#define CH 64
#define ZF 160   // 2C + D (=144) padded to 5 K-tiles of 32
#define KF 128   // node-update K (x | aggr)

// ---------------------------------------------------------------- zero scratch
__global__ void zero_kernel(float* __restrict__ p, int n) {
    int i = blockIdx.x * blockDim.x + threadIdx.x;
    if (i < n) p[i] = 0.0f;
}

// ---------------------------------------------------------------- degree count
__global__ void count_kernel(const int* __restrict__ ei, float* __restrict__ cnt, int E) {
    int e = blockIdx.x * blockDim.x + threadIdx.x;
    if (e < E) atomicAdd(&cnt[ei[E + e]], 1.0f);
}

// ---------------------------------------------------------------- edge messages
// Block = 256 threads = 8 waves = 2 edge M-tiles x 4 channel N-tiles.
// Per 32-edge iteration: indices staged to LDS, z = [x_dst|x_src|e|0] staged
// as bf16, then each wave runs 5 v_wmma_f32_16x16x32_bf16 and atomically
// scatters its 16x16 message tile into sums[dst].
__global__ void __launch_bounds__(256)
edge_msg_kernel(const float* __restrict__ x,
                const int*   __restrict__ ei,   // [2,E] row0=src, row1=dst
                const float* __restrict__ ea,   // [E,16]
                const float* __restrict__ gW,   // [144,64] row-major
                const float* __restrict__ gb,   // [64]
                float* __restrict__ sums,       // [N,64]
                int N, int E) {
    __shared__ __align__(16) unsigned short wlds[CH * ZF];  // col-major [c][f]
    __shared__ __align__(16) unsigned short zlds[32 * ZF];  // [m][f], 2 M-tiles
    __shared__ __align__(16) int srclds[32];
    __shared__ __align__(16) int dstlds[32];

    const int tid   = threadIdx.x;
    const int lane  = tid & 31;
    const int wave  = tid >> 5;
    const int n0    = (wave & 3) * 16;   // channel tile
    const int mt    = wave >> 2;         // edge sub-tile (0 or 1)
    const int mbase = mt * 16;

    // g_W -> LDS, column-major, bf16, zero-padded K 144..159
    for (int i = tid; i < CH * ZF; i += 256) {
        int cc = i / ZF, f = i % ZF;
        float v = (f < 2 * CH + 16) ? gW[f * CH + cc] : 0.0f;
        wlds[i] = f32_to_bf16(v);
    }
    __syncthreads();

    // Build the 5 B fragments for this wave's channels (live across all tiles).
    BF16x16 bfrag[5];
    {
        const int n  = n0 + (lane & 15);
        const int kb = (lane >> 4) << 3;   // 0 or 8
        #pragma unroll
        for (int t = 0; t < 5; ++t) {
            const unsigned short* p = &wlds[n * ZF + t * 32 + kb];
            bfrag[t].h[0] = *(const ushort8*)(p);
            bfrag[t].h[1] = *(const ushort8*)(p + 16);
        }
    }

    const int ntiles = (E + 31) >> 5;    // 32 edges per block iteration
    for (int tile = blockIdx.x; tile < ntiles; tile += gridDim.x) {
        const int  e0   = tile << 5;
        const bool full = (e0 + 32 <= E);
        __syncthreads();                 // protect zlds/index reuse
        if (tid == 0 && e0 + 32 < E)
            __builtin_prefetch(&ei[E + e0 + 32], 0, 0);   // global_prefetch_b8

        // Stage this iteration's edge indices once.
        if (tid < 32) {
            int e = e0 + tid;
            int s = 0, d = 0;
            if (full || e < E) { s = ei[e]; d = ei[E + e]; }
            srclds[tid] = s;
            dstlds[tid] = d;
        }
        __syncthreads();

        // Stage z tile (32 edges x 160 feats) as bf16, indices from LDS.
        for (int i = tid; i < 32 * 40; i += 256) {
            int m = i / 40, q = (i % 40) << 2;   // 4-feature group
            float4 v = make_float4(0.f, 0.f, 0.f, 0.f);
            if (full || (e0 + m) < E) {
                if (q < CH) {
                    v = *(const float4*)&x[(size_t)dstlds[m] * CH + q];
                } else if (q < 2 * CH) {
                    v = *(const float4*)&x[(size_t)srclds[m] * CH + (q - CH)];
                } else if (q < 2 * CH + 16) {
                    v = *(const float4*)&ea[(size_t)(e0 + m) * 16 + (q - 2 * CH)];
                }
            }
            unsigned short* zp = &zlds[m * ZF + q];
            zp[0] = f32_to_bf16(v.x);
            zp[1] = f32_to_bf16(v.y);
            zp[2] = f32_to_bf16(v.z);
            zp[3] = f32_to_bf16(v.w);
        }
        __syncthreads();

        // 5 K-steps of bf16 WMMA, fp32 accumulate.
        v8f acc = {0.f, 0.f, 0.f, 0.f, 0.f, 0.f, 0.f, 0.f};
        const int m    = mbase + (lane & 15);
        const int koff = (lane >> 4) << 3;   // 0 or 8
        #pragma unroll
        for (int t = 0; t < 5; ++t) {
            BF16x16 a;
            const unsigned short* p = &zlds[m * ZF + t * 32 + koff];
            a.h[0] = *(const ushort8*)(p);        // K koff..koff+7
            a.h[1] = *(const ushort8*)(p + 16);   // K koff+16..koff+23
            acc = __builtin_amdgcn_wmma_f32_16x16x32_bf16(
                false, a.v, false, bfrag[t].v, (short)0, acc, false, false);
        }

        // Scatter (+bias). C/D layout: lane -> channel, elem r -> row r+8*(lane>=16).
        const int   cc   = n0 + (lane & 15);
        const float bias = gb[cc];
        const int   rb   = mbase + ((lane >> 4) << 3);   // first edge row for elem 0
        int4 d0 = *(const int4*)&dstlds[rb];             // 8 dst ids via 2 b128 ds reads
        int4 d1 = *(const int4*)&dstlds[rb + 4];
        int  dsts[8] = {d0.x, d0.y, d0.z, d0.w, d1.x, d1.y, d1.z, d1.w};
        float* sc = sums + cc;
        if (full) {
            #pragma unroll
            for (int r = 0; r < 8; ++r)
                atomicAdd(&sc[(size_t)dsts[r] * CH], acc[r] + bias);
        } else {
            #pragma unroll
            for (int r = 0; r < 8; ++r)
                if (e0 + rb + r < E)
                    atomicAdd(&sc[(size_t)dsts[r] * CH], acc[r] + bias);
        }
    }
}

// ---------------------------------------------------------------- node update
__global__ void __launch_bounds__(128)
node_update_kernel(const float* __restrict__ x,
                   const float* __restrict__ sums,
                   const float* __restrict__ cnt,
                   const float* __restrict__ fW,    // [128,64] row-major
                   const float* __restrict__ fb,    // [64]
                   const float* __restrict__ clsW,  // [64]
                   const float* __restrict__ clsb,  // [1]
                   float* __restrict__ out,         // [2*N]: y then probs
                   int N) {
    __shared__ __align__(16) unsigned short flds[CH * KF];  // col-major [c][f]
    __shared__ __align__(16) unsigned short zin[16 * KF];   // [m][f]
    __shared__ float hlds[16 * CH];

    const int tid  = threadIdx.x;
    const int lane = tid & 31;
    const int wave = tid >> 5;
    const int n0   = wave * 16;

    for (int i = tid; i < CH * KF; i += 128) {
        int cc = i / KF, f = i % KF;
        flds[i] = f32_to_bf16(fW[f * CH + cc]);
    }
    __syncthreads();

    BF16x16 bfrag[4];
    {
        const int n  = n0 + (lane & 15);
        const int kb = (lane >> 4) << 3;
        #pragma unroll
        for (int t = 0; t < 4; ++t) {
            const unsigned short* p = &flds[n * KF + t * 32 + kb];
            bfrag[t].h[0] = *(const ushort8*)(p);
            bfrag[t].h[1] = *(const ushort8*)(p + 16);
        }
    }

    const int ntiles = (N + 15) >> 4;
    for (int tile = blockIdx.x; tile < ntiles; tile += gridDim.x) {
        const int  v0   = tile << 4;
        const bool full = (v0 + 16 <= N);
        __syncthreads();

        // Stage [x | aggr] (16 nodes x 128) as bf16; aggr = sums / max(cnt,1).
        for (int i = tid; i < 16 * 32; i += 128) {
            int m = i / 32, q = (i % 32) << 2;
            int v = v0 + m;
            float4 val = make_float4(0.f, 0.f, 0.f, 0.f);
            if (full || v < N) {
                if (q < CH) {
                    val = *(const float4*)&x[(size_t)v * CH + q];
                } else {
                    float  inv = 1.0f / fmaxf(cnt[v], 1.0f);
                    float4 s   = *(const float4*)&sums[(size_t)v * CH + (q - CH)];
                    val = make_float4(s.x * inv, s.y * inv, s.z * inv, s.w * inv);
                }
            }
            unsigned short* zp = &zin[m * KF + q];
            zp[0] = f32_to_bf16(val.x);
            zp[1] = f32_to_bf16(val.y);
            zp[2] = f32_to_bf16(val.z);
            zp[3] = f32_to_bf16(val.w);
        }
        __syncthreads();

        v8f acc = {0.f, 0.f, 0.f, 0.f, 0.f, 0.f, 0.f, 0.f};
        const int m    = lane & 15;
        const int koff = (lane >> 4) << 3;
        #pragma unroll
        for (int t = 0; t < 4; ++t) {
            BF16x16 a;
            const unsigned short* p = &zin[m * KF + t * 32 + koff];
            a.h[0] = *(const ushort8*)(p);
            a.h[1] = *(const ushort8*)(p + 16);
            acc = __builtin_amdgcn_wmma_f32_16x16x32_bf16(
                false, a.v, false, bfrag[t].v, (short)0, acc, false, false);
        }

        // h (+bias) to LDS for the classifier reduction.
        const int   cc    = n0 + (lane & 15);
        const float bias  = fb[cc];
        const int   mbase = (lane >> 4) << 3;
        #pragma unroll
        for (int r = 0; r < 8; ++r)
            hlds[(mbase + r) * CH + cc] = acc[r] + bias;
        __syncthreads();

        if (tid < 16) {
            int v = v0 + tid;
            if (full || v < N) {
                float s = clsb[0];
                #pragma unroll 8
                for (int c = 0; c < CH; ++c) s += hlds[tid * CH + c] * clsW[c];
                float prob  = 1.0f / (1.0f + __expf(-s));
                out[N + v]  = prob;
                out[v]      = (prob > 0.5f) ? 1.0f : 0.0f;
            }
        }
        // top-of-loop barrier orders hlds reads before next iteration's writes
    }
}

extern "C" void kernel_launch(void* const* d_in, const int* in_sizes, int n_in,
                              void* d_out, int out_size, void* d_ws, size_t ws_size,
                              hipStream_t stream) {
    const float* x    = (const float*)d_in[0];
    const int*   ei   = (const int*)  d_in[1];
    const float* ea   = (const float*)d_in[2];
    const float* gW   = (const float*)d_in[3];
    const float* gb   = (const float*)d_in[4];
    const float* fW   = (const float*)d_in[5];
    const float* fb   = (const float*)d_in[6];
    const float* clsW = (const float*)d_in[7];
    const float* clsb = (const float*)d_in[8];
    float*       out  = (float*)d_out;

    const int N = in_sizes[0] / CH;
    const int E = in_sizes[1] / 2;

    float* sums = (float*)d_ws;            // [N,64]
    float* cnt  = sums + (size_t)N * CH;   // [N]

    const int zn = N * CH + N;
    zero_kernel<<<(zn + 255) / 256, 256, 0, stream>>>(sums, zn);
    count_kernel<<<(E + 255) / 256, 256, 0, stream>>>(ei, cnt, E);

    int etiles = (E + 31) / 32;
    int egrid  = etiles < 4096 ? etiles : 4096;
    edge_msg_kernel<<<egrid, 256, 0, stream>>>(x, ei, ea, gW, gb, sums, N, E);

    int ntl   = (N + 15) / 16;
    int ngrid = ntl < 1024 ? ntl : 1024;
    node_update_kernel<<<ngrid, 128, 0, stream>>>(x, sums, cnt, fW, fb,
                                                  clsW, clsb, out, N);
}
